// TemporalMHA_4518305595530
// MI455X (gfx1250) — matrix operand order repaired and checked by reading, not compiled, not run on hardware
//
#include <hip/hip_runtime.h>

// ---------------- problem constants ----------------
#define T_LEN   512
#define BK_N    64
#define D_MODEL 512
#define H_N     8
#define DH      64            // v head dim
#define QKD     128           // qk head dim (concat dec+time)
#define QT      128           // query tile per block (16 rows per wave, 8 waves)
#define KTILE   128           // key tile per iteration
#define LDK     136           // padded LDS row stride (elements): 272B rows -> conflict-free
#define BKD     (BK_N * D_MODEL)

typedef __attribute__((ext_vector_type(16))) __bf16 v16bf;
typedef __attribute__((ext_vector_type(8)))  float  v8f;

union FragBF {
    v16bf v;
    uint4 q[2];
};

// f32 -> bf16, manual round-to-nearest-even
__device__ __forceinline__ unsigned short f2bf(float f) {
    union { float f; unsigned int u; } x; x.f = f;
    unsigned int u = x.u;
    u += 0x7FFFu + ((u >> 16) & 1u);
    return (unsigned short)(u >> 16);
}
__device__ __forceinline__ unsigned int pack2bf(float a, float b) {
    return (unsigned int)f2bf(a) | ((unsigned int)f2bf(b) << 16);
}

// raw v_exp_f32 (base-2); FTZ below 2^-126 is fine for softmax weights
__device__ __forceinline__ float fast_exp2(float x) {
    return __builtin_amdgcn_exp2f(x);
}

// A-matrix fragment (16x32 bf16), ISA 7.12.2: lane(0..15)=M, halves split K;
// u[0..3] -> K = half*8 + 0..7 ; u[4..7] -> K = 16 + half*8 + 0..7
__device__ __forceinline__ v16bf load_afrag(const unsigned short* base, int row, int k0, int hgrp) {
    FragBF r;
    const unsigned short* p = base + row * LDK + k0 + hgrp * 8;
    r.q[0] = *(const uint4*)(p);
    r.q[1] = *(const uint4*)(p + 16);
    return r.v;
}

// B-matrix fragment (32x16 bf16): lane%16 = N, halves split K like A;
// u[0..3] -> K = half*8 + 0..7 ; u[4..7] -> K = 16 + half*8 + 0..7
__device__ __forceinline__ v16bf load_bfrag(const unsigned short* base, int n0, int k0, int nl, int hgrp) {
    FragBF r;
    const unsigned short* p = base + (n0 + nl) * LDK + k0 + hgrp * 8;
    r.q[0] = *(const uint4*)(p);
    r.q[1] = *(const uint4*)(p + 16);
    return r.v;
}

__global__ __launch_bounds__(256) void TemporalMHA_fa_kernel(
        const float* __restrict__ dec,
        const float* __restrict__ tpe,
        float* __restrict__ out) {

    // LDS: K tile (128 x 128 bf16, padded), Q-staging / P buffer (128 x 128), V^T (64 x 128)
    __shared__ __align__(16) unsigned short sK [KTILE * LDK];
    __shared__ __align__(16) unsigned short sQP[QT    * LDK];
    __shared__ __align__(16) unsigned short sVt[DH    * LDK];

    const int tid  = threadIdx.x;
    const int wv   = tid >> 5;
    const int lane = tid & 31;
    const int hgrp = lane >> 4;   // 0: lanes 0-15, 1: lanes 16-31
    const int nl   = lane & 15;

    const int qb = blockIdx.x & 3;         // query tile index (T/QT = 4)
    const int bh = blockIdx.x >> 2;
    const int h  = bh & (H_N - 1);
    const int bk = bh >> 3;
    const int q0 = qb * QT;

    // base-2 softmax domain: fold 1/sqrt(128) * log2(e) into one constant
    const float scale2 = 0.08838834764831845f * 1.4426950408889634f;
    const size_t colbase = (size_t)bk * D_MODEL + (size_t)h * DH;

    // ---- stage Q tile: global f32 -> bf16 LDS [row][0:64)=dec, [64:128)=time ----
    for (int idx = tid; idx < QT * 32; idx += 256) {
        int row = idx >> 5, c4 = idx & 31;
        int isDec = (c4 < 16);
        int c = isDec ? c4 : c4 - 16;
        const float* src = isDec ? dec : tpe;
        const float4 v = *(const float4*)(src + (size_t)(q0 + row) * BKD + colbase + c * 4);
        int col = (isDec ? 0 : 64) + c * 4;
        *(unsigned int*)&sQP[row * LDK + col]     = pack2bf(v.x, v.y);
        *(unsigned int*)&sQP[row * LDK + col + 2] = pack2bf(v.z, v.w);
    }
    __syncthreads();

    // per-wave Q fragments: 16 rows x 128 K, as 4 A-frags
    v16bf qf[4];
#pragma unroll
    for (int kt = 0; kt < 4; ++kt)
        qf[kt] = load_afrag(sQP, wv * 16 + nl, kt * 32, hgrp);
    __syncthreads();   // sQP now reusable as per-wave P buffer

    // accumulators: O (16x64) in C/D layout, online-softmax stats per row r (log2 domain)
    v8f o[4];
    float m_i[8], l_i[8];
#pragma unroll
    for (int vt = 0; vt < 4; ++vt) o[vt] = (v8f){0,0,0,0,0,0,0,0};
#pragma unroll
    for (int r = 0; r < 8; ++r) { m_i[r] = -1e30f; l_i[r] = 0.0f; }

    for (int kb = 0; kb < T_LEN / KTILE; ++kb) {
        const int k0 = kb * KTILE;

        // ---- stage K tile (and V^T from its dec half) ----
        for (int idx = tid; idx < KTILE * 32; idx += 256) {
            int row = idx >> 5, c4 = idx & 31;
            int isDec = (c4 < 16);
            int c = isDec ? c4 : c4 - 16;
            const float* src = isDec ? dec : tpe;
            const float4 v = *(const float4*)(src + (size_t)(k0 + row) * BKD + colbase + c * 4);
            int col = (isDec ? 0 : 64) + c * 4;
            *(unsigned int*)&sK[row * LDK + col]     = pack2bf(v.x, v.y);
            *(unsigned int*)&sK[row * LDK + col + 2] = pack2bf(v.z, v.w);
            if (isDec) {       // V^T[d][key] = V[key][d]
                int d0 = c * 4;
                sVt[(d0 + 0) * LDK + row] = f2bf(v.x);
                sVt[(d0 + 1) * LDK + row] = f2bf(v.y);
                sVt[(d0 + 2) * LDK + row] = f2bf(v.z);
                sVt[(d0 + 3) * LDK + row] = f2bf(v.w);
            }
        }
        // ---- prefetch next K tile into L2/L0 (global_prefetch_b8) ----
        if (kb + 1 < T_LEN / KTILE) {
            int prow = tid >> 1;             // 128 rows x 2 sources, 1 thread each
            const float* psrc = (tid & 1) ? tpe : dec;
            const float* pa = psrc + (size_t)(k0 + KTILE + prow) * BKD + colbase;
            __builtin_prefetch(pa, 0, 1);        // first 128B line of the 256B row segment
            __builtin_prefetch(pa + 32, 0, 1);   // second 128B line
        }
        __syncthreads();

        // ---- S = Q K^T : 8 key sub-tiles x 4 K-steps = 32 WMMA ----
        v8f st[8];
#pragma unroll
        for (int nt = 0; nt < 8; ++nt) {
            v8f acc = (v8f){0,0,0,0,0,0,0,0};
#pragma unroll
            for (int kt = 0; kt < 4; ++kt) {
                v16bf bf = load_bfrag(sK, nt * 16, kt * 32, nl, hgrp);
                acc = __builtin_amdgcn_wmma_f32_16x16x32_bf16(
                          false, qf[kt], false, bf, (short)0, acc, false, false);
            }
            st[nt] = acc;
        }

        // ---- online softmax in log2 domain (rows m = r + 8*hgrp) ----
        float nmx[8], alpha[8], rs[8];
#pragma unroll
        for (int r = 0; r < 8; ++r) {
            float mx = -1e30f;
#pragma unroll
            for (int nt = 0; nt < 8; ++nt) mx = fmaxf(mx, st[nt][r]);
            mx = fmaxf(mx, __shfl_xor(mx, 1, 32));
            mx = fmaxf(mx, __shfl_xor(mx, 2, 32));
            mx = fmaxf(mx, __shfl_xor(mx, 4, 32));
            mx = fmaxf(mx, __shfl_xor(mx, 8, 32));
            mx *= scale2;
            float nm = fmaxf(m_i[r], mx);
            nmx[r]   = nm;
            alpha[r] = fast_exp2(m_i[r] - nm);
            m_i[r]   = nm;
            rs[r]    = 0.0f;
        }

        // ---- P = 2^(S*scale2 - m), C-layout -> LDS (per-wave private strip) ----
#pragma unroll
        for (int nt = 0; nt < 8; ++nt) {
#pragma unroll
            for (int r = 0; r < 8; ++r) {
                float p = fast_exp2(__builtin_fmaf(st[nt][r], scale2, -nmx[r]));
                rs[r] += p;
                sQP[(wv * 16 + r + 8 * hgrp) * LDK + nt * 16 + nl] = f2bf(p);
            }
        }
#pragma unroll
        for (int r = 0; r < 8; ++r) {
            float s = rs[r];
            s += __shfl_xor(s, 1, 32);
            s += __shfl_xor(s, 2, 32);
            s += __shfl_xor(s, 4, 32);
            s += __shfl_xor(s, 8, 32);
            l_i[r] = l_i[r] * alpha[r] + s;
        }
#pragma unroll
        for (int vt = 0; vt < 4; ++vt)
#pragma unroll
            for (int r = 0; r < 8; ++r)
                o[vt][r] *= alpha[r];

        // LDS ops from one wave are in-order (DScnt in-order); just pin ordering
        __builtin_amdgcn_wave_barrier();

        // ---- O += P V : re-read P as A-frags, V^T gives contiguous B-frags ----
        v16bf pf[4];
#pragma unroll
        for (int kt = 0; kt < 4; ++kt)
            pf[kt] = load_afrag(sQP, wv * 16 + nl, kt * 32, hgrp);

#pragma unroll
        for (int vt = 0; vt < 4; ++vt) {
            v8f acc = o[vt];
#pragma unroll
            for (int kt = 0; kt < 4; ++kt) {
                v16bf bv = load_bfrag(sVt, vt * 16, kt * 32, nl, hgrp);
                acc = __builtin_amdgcn_wmma_f32_16x16x32_bf16(
                          false, pf[kt], false, bv, (short)0, acc, false, false);
            }
            o[vt] = acc;
        }
        __syncthreads();   // protect sK/sVt before next tile load
    }

    // ---- epilogue: O / l, scatter back to [T, BK, D]; non-temporal (write-once) ----
#pragma unroll
    for (int r = 0; r < 8; ++r) {
        float inv = 1.0f / l_i[r];
        int qr = q0 + wv * 16 + r + 8 * hgrp;
#pragma unroll
        for (int vt = 0; vt < 4; ++vt)
            __builtin_nontemporal_store(o[vt][r] * inv,
                &out[(size_t)qr * BKD + colbase + vt * 16 + nl]);
    }
}

extern "C" void kernel_launch(void* const* d_in, const int* in_sizes, int n_in,
                              void* d_out, int out_size, void* d_ws, size_t ws_size,
                              hipStream_t stream) {
    const float* dec = (const float*)d_in[0];
    const float* tpe = (const float*)d_in[1];
    float* out = (float*)d_out;
    (void)in_sizes; (void)n_in; (void)out_size; (void)d_ws; (void)ws_size;

    dim3 grid((T_LEN / QT) * BK_N * H_N);   // 4 * 64 * 8 = 2048 blocks
    dim3 block(256);                        // 8 waves of 32
    TemporalMHA_fa_kernel<<<grid, block, 0, stream>>>(dec, tpe, out);
}
